// FastGAT_83193516524089
// MI455X (gfx1250) — matrix operand order repaired and checked by reading, not compiled, not run on hardware
//
#include <hip/hip_runtime.h>

typedef __attribute__((ext_vector_type(16))) _Float16 v16h;
typedef __attribute__((ext_vector_type(8)))  float    v8f;

#define HID 128
#define HEADS 4
#define CH 32

__device__ __forceinline__ float lrelu(float x, float s) { return x > 0.f ? x : s * x; }

// IEEE-ordered float atomic max via signed/unsigned int punning (init with -inf).
__device__ __forceinline__ void atomicMaxFloat(float* addr, float val) {
    if (val >= 0.f) atomicMax((int*)addr, __float_as_int(val));
    else            atomicMin((unsigned int*)addr, __float_as_uint(val));
}

// ---------------------------------------------------------------------------
// WMMA GEMM: out[N, NCOLS] = epilogue(A[N, KDIM] @ W[KDIM, NCOLS] + bias)
// MODE 0: +bias                      (lin_l / lin_r)
// MODE 1: +bias -> BN -> lrelu(0.1)  (pre layer)
// MODE 2: +bias -> lrelu(0.1)        (head first linear)
// Block: 32 * (NCOLS/16) threads (1 wave per 16-col tile); 128 rows per block.
// Tail rows are CLAMPED on load (unconditional global_load_b128) and the store
// epilogue branches on a UNIFORM full-subtile test so only the final partial
// block pays per-element predication.
// ---------------------------------------------------------------------------
template <int MODE, int KDIM, int NCOLS>
__global__ void gemm_wmma(const float* __restrict__ A, int nRows,
                          const float* __restrict__ W,
                          const float* __restrict__ bias,
                          const float* __restrict__ bn_g, const float* __restrict__ bn_b,
                          const float* __restrict__ bn_m, const float* __restrict__ bn_v,
                          float* __restrict__ out) {
    const int lane = threadIdx.x & 31;
    const int wave = threadIdx.x >> 5;
    const int col  = wave * 16 + (lane & 15);
    const int r0b  = blockIdx.x * 128;
    constexpr int NK = KDIM / 32;

    // per-lane (per-column) epilogue constants
    float bias_c = bias[col];
    float scale_c = 0.f, shift_c = 0.f, mean_c = 0.f;
    if (MODE == 1) {
        mean_c  = bn_m[col];
        scale_c = bn_g[col] * rsqrtf(bn_v[col] + 1e-5f);
        shift_c = bn_b[col];
    }

    // Preload all B fragments (weights); kept in VGPRs and reused across the
    // 8 row-subtiles.  B 32x16 f16 layout: elems 0..7 -> K=kb2+0..7,
    // elems 8..15 -> K=16+kb2+0..7, kb2 = 8*(lane>=16), column = lane&15.
    const int kb2 = (lane >> 4) ? 8 : 0;
    v16h bf[NK];
#pragma unroll
    for (int kk = 0; kk < NK; ++kk) {
        const float* wp = W + (size_t)(kk * 32 + kb2) * NCOLS + col;
        v16h b;
#pragma unroll
        for (int i = 0; i < 16; ++i) b[i] = (_Float16)wp[(size_t)(i < 8 ? i : i + 8) * NCOLS];
        bf[kk] = b;
    }

    // A 16x32 f16 layout: row = lane&15; elems 0..7 -> K=kbA+0..7,
    // elems 8..15 -> K=16+kbA+0..7, kbA = 8*(lane>=16).
    const int kbA = (lane >> 4) ? 8 : 0;
#pragma unroll
    for (int st = 0; st < 8; ++st) {
        int row = r0b + st * 16 + (lane & 15);
        row = row < nRows ? row : nRows - 1;          // clamp: no predicated loads
        const float* ap = A + (size_t)row * KDIM;
        v8f acc = {};
#pragma unroll
        for (int kk = 0; kk < NK; ++kk) {
            const float* apk = ap + kk * 32 + kbA;    // 16B aligned
            const float4 x0 = ((const float4*)apk)[0];
            const float4 x1 = ((const float4*)apk)[1];
            const float4 y0 = ((const float4*)(apk + 16))[0];
            const float4 y1 = ((const float4*)(apk + 16))[1];
            v16h a;
            a[0]  = (_Float16)x0.x; a[1]  = (_Float16)x0.y;
            a[2]  = (_Float16)x0.z; a[3]  = (_Float16)x0.w;
            a[4]  = (_Float16)x1.x; a[5]  = (_Float16)x1.y;
            a[6]  = (_Float16)x1.z; a[7]  = (_Float16)x1.w;
            a[8]  = (_Float16)y0.x; a[9]  = (_Float16)y0.y;
            a[10] = (_Float16)y0.z; a[11] = (_Float16)y0.w;
            a[12] = (_Float16)y1.x; a[13] = (_Float16)y1.y;
            a[14] = (_Float16)y1.z; a[15] = (_Float16)y1.w;
            acc = __builtin_amdgcn_wmma_f32_16x16x32_f16(false, a, false, bf[kk],
                                                         (short)0, acc, false, false);
        }
        // C/D f32 16x16: lanes 0-15 -> M=v, lanes 16-31 -> M=8+v, N=lane&15.
        const int rowbase = r0b + st * 16 + ((lane >> 4) ? 8 : 0);
        float res[8];
#pragma unroll
        for (int v = 0; v < 8; ++v) {
            float val = acc[v] + bias_c;
            if (MODE == 1) {
                val = (val - mean_c) * scale_c + shift_c;
                val = lrelu(val, 0.1f);
            } else if (MODE == 2) {
                val = lrelu(val, 0.1f);
            }
            res[v] = val;
        }
        float* op = out + (size_t)rowbase * NCOLS + col;
        if (r0b + (st + 1) * 16 <= nRows) {           // uniform: full subtile
#pragma unroll
            for (int v = 0; v < 8; ++v) op[(size_t)v * NCOLS] = res[v];
        } else {                                      // tail subtile only
#pragma unroll
            for (int v = 0; v < 8; ++v)
                if (rowbase + v < nRows) op[(size_t)v * NCOLS] = res[v];
        }
    }
}

// --------------------------- edge-phase kernels ----------------------------

__global__ void k_init(float* __restrict__ AGG, float* __restrict__ MX,
                       float* __restrict__ DEN, int N) {
    int t = blockIdx.x * blockDim.x + threadIdx.x;
    if (t >= N * HID) return;
    AGG[t] = 0.f;
    int c = t & (HID - 1);
    if (c < HEADS) {
        int n = t >> 7;
        MX[n * HEADS + c]  = -__builtin_inff();
        DEN[n * HEADS + c] = 0.f;
    }
}

// logits[e,h] = sum_c att[h,c] * lrelu(xr[dst,h,c] + xl[src,h,c], 0.2);  segment max.
__global__ void k_logit(const float* __restrict__ XL, const float* __restrict__ XR,
                        const float* __restrict__ att, const int* __restrict__ ei,
                        int E, int E2, float* __restrict__ EB, float* __restrict__ MX) {
    int t = blockIdx.x * blockDim.x + threadIdx.x;
    if (t >= E2 * HEADS) return;
    int h = t & (HEADS - 1), e = t >> 2;
    int src, dst;
    if (e < E) { src = ei[e]; dst = ei[E + e]; } else { src = dst = e - E; }
    const float4* pl = (const float4*)(XL + (size_t)src * HID + h * CH);
    const float4* pr = (const float4*)(XR + (size_t)dst * HID + h * CH);
    const float4* pa = (const float4*)(att + h * CH);
    float acc = 0.f;
#pragma unroll
    for (int i = 0; i < 8; ++i) {
        float4 a = pl[i], b = pr[i], w = pa[i];
        float s;
        s = a.x + b.x; acc += lrelu(s, 0.2f) * w.x;
        s = a.y + b.y; acc += lrelu(s, 0.2f) * w.y;
        s = a.z + b.z; acc += lrelu(s, 0.2f) * w.z;
        s = a.w + b.w; acc += lrelu(s, 0.2f) * w.w;
    }
    EB[t] = acc;
    atomicMaxFloat(&MX[(size_t)dst * HEADS + h], acc);
}

// EB <- exp(logit - mx[dst]);  DEN[dst] += EB
__global__ void k_expsum(float* __restrict__ EB, const float* __restrict__ MX,
                         float* __restrict__ DEN, const int* __restrict__ ei,
                         int E, int E2) {
    int t = blockIdx.x * blockDim.x + threadIdx.x;
    if (t >= E2 * HEADS) return;
    int h = t & (HEADS - 1), e = t >> 2;
    int dst = (e < E) ? ei[E + e] : (e - E);
    float v = __expf(EB[t] - MX[(size_t)dst * HEADS + h]);
    EB[t] = v;
    atomicAdd(&DEN[(size_t)dst * HEADS + h], v);
}

// AGG[dst, ch] += alpha * XL[src, ch]
__global__ void k_scatter(const float* __restrict__ XL, const float* __restrict__ EB,
                          const float* __restrict__ DEN, const int* __restrict__ ei,
                          int E, int E2, float* __restrict__ AGG) {
    long long t = (long long)blockIdx.x * blockDim.x + threadIdx.x;
    if (t >= (long long)E2 * HID) return;
    int ch = (int)(t & (HID - 1));
    long long e = t >> 7;
    int h = ch >> 5;
    int src, dst;
    if (e < E) { src = ei[e]; dst = ei[E + e]; } else { src = dst = (int)(e - E); }
    float alpha = EB[e * HEADS + h] / (DEN[(size_t)dst * HEADS + h] + 1e-16f);
    atomicAdd(&AGG[(size_t)dst * HID + ch], alpha * XL[(size_t)src * HID + ch]);
}

// out = lrelu01(BN(AGG + attn_bias)) + RES
__global__ void k_post(const float* __restrict__ AGG, const float* __restrict__ bias,
                       const float* __restrict__ g, const float* __restrict__ be,
                       const float* __restrict__ m, const float* __restrict__ v,
                       const float* __restrict__ RES, float* __restrict__ OUT, int N) {
    int t = blockIdx.x * blockDim.x + threadIdx.x;
    if (t >= N * HID) return;
    int c = t & (HID - 1);
    float val = AGG[t] + bias[c];
    val = (val - m[c]) * (g[c] * rsqrtf(v[c] + 1e-5f)) + be[c];
    OUT[t] = lrelu(val, 0.1f) + RES[t];
}

// out[i] = dot(T[i, 0:64], Wo2) + bo2
__global__ void k_gemv(const float* __restrict__ T, const float* __restrict__ W2,
                       const float* __restrict__ b2, float* __restrict__ out, int N) {
    int i = blockIdx.x * blockDim.x + threadIdx.x;
    if (i >= N) return;
    const float4* tp = (const float4*)(T + (size_t)i * 64);
    const float4* wp = (const float4*)W2;
    float acc = b2[0];
#pragma unroll
    for (int j = 0; j < 16; ++j) {
        float4 a = tp[j], w = wp[j];
        acc += a.x * w.x + a.y * w.y + a.z * w.z + a.w * w.w;
    }
    out[i] = acc;
}

// ---------------------------------------------------------------------------

extern "C" void kernel_launch(void* const* d_in, const int* in_sizes, int n_in,
                              void* d_out, int out_size, void* d_ws, size_t ws_size,
                              hipStream_t stream) {
    const float* x     = (const float*)d_in[0];
    const int*   ei    = (const int*)d_in[1];
    const float* W_pre = (const float*)d_in[2];
    const float* b_pre = (const float*)d_in[3];
    const float* g_pre = (const float*)d_in[4];
    const float* be_pre= (const float*)d_in[5];
    const float* m_pre = (const float*)d_in[6];
    const float* v_pre = (const float*)d_in[7];
    const float* Wl1 = (const float*)d_in[8];  const float* bl1 = (const float*)d_in[9];
    const float* Wr1 = (const float*)d_in[10]; const float* br1 = (const float*)d_in[11];
    const float* att1= (const float*)d_in[12]; const float* bias1=(const float*)d_in[13];
    const float* g1  = (const float*)d_in[14]; const float* be1 = (const float*)d_in[15];
    const float* m1  = (const float*)d_in[16]; const float* v1  = (const float*)d_in[17];
    const float* Wl2 = (const float*)d_in[18]; const float* bl2 = (const float*)d_in[19];
    const float* Wr2 = (const float*)d_in[20]; const float* br2 = (const float*)d_in[21];
    const float* att2= (const float*)d_in[22]; const float* bias2=(const float*)d_in[23];
    const float* g2  = (const float*)d_in[24]; const float* be2 = (const float*)d_in[25];
    const float* m2  = (const float*)d_in[26]; const float* v2  = (const float*)d_in[27];
    const float* Wo1 = (const float*)d_in[28]; const float* bo1 = (const float*)d_in[29];
    const float* Wo2 = (const float*)d_in[30]; const float* bo2 = (const float*)d_in[31];

    const int N  = in_sizes[0] / 64;
    const int E  = in_sizes[1] / 2;
    const int E2 = E + N;

    float* ws = (float*)d_ws;
    size_t o = 0;
    float* H   = ws + o; o += (size_t)N * HID;
    float* H2  = ws + o; o += (size_t)N * HID;
    float* XL  = ws + o; o += (size_t)N * HID;
    float* XR  = ws + o; o += (size_t)N * HID;
    float* AGG = ws + o; o += (size_t)N * HID;
    float* EB  = ws + o; o += (size_t)E2 * HEADS;
    float* MX  = ws + o; o += (size_t)N * HEADS;
    float* DEN = ws + o; o += (size_t)N * HEADS;

    const dim3 gRow((N + 127) / 128);
    const int tN  = N * HID;
    const int tEH = E2 * HEADS;
    const long long tEC = (long long)E2 * HID;
    const int bN  = (tN + 255) / 256;
    const int bEH = (tEH + 255) / 256;
    const int bEC = (int)((tEC + 255) / 256);
    const int bGV = (N + 255) / 256;

    // pre: H = lrelu01(BN(x @ W_pre + b_pre))
    gemm_wmma<1, 64, HID><<<gRow, 256, 0, stream>>>(x, N, W_pre, b_pre,
                                                    g_pre, be_pre, m_pre, v_pre, H);
    // ---- GAT layer 1 ----
    gemm_wmma<0, HID, HID><<<gRow, 256, 0, stream>>>(H, N, Wl1, bl1, 0, 0, 0, 0, XL);
    gemm_wmma<0, HID, HID><<<gRow, 256, 0, stream>>>(H, N, Wr1, br1, 0, 0, 0, 0, XR);
    k_init   <<<bN, 256, 0, stream>>>(AGG, MX, DEN, N);
    k_logit  <<<bEH, 256, 0, stream>>>(XL, XR, att1, ei, E, E2, EB, MX);
    k_expsum <<<bEH, 256, 0, stream>>>(EB, MX, DEN, ei, E, E2);
    k_scatter<<<bEC, 256, 0, stream>>>(XL, EB, DEN, ei, E, E2, AGG);
    k_post   <<<bN, 256, 0, stream>>>(AGG, bias1, g1, be1, m1, v1, H, H2, N);

    // ---- GAT layer 2 ----
    gemm_wmma<0, HID, HID><<<gRow, 256, 0, stream>>>(H2, N, Wl2, bl2, 0, 0, 0, 0, XL);
    gemm_wmma<0, HID, HID><<<gRow, 256, 0, stream>>>(H2, N, Wr2, br2, 0, 0, 0, 0, XR);
    k_init   <<<bN, 256, 0, stream>>>(AGG, MX, DEN, N);
    k_logit  <<<bEH, 256, 0, stream>>>(XL, XR, att2, ei, E, E2, EB, MX);
    k_expsum <<<bEH, 256, 0, stream>>>(EB, MX, DEN, ei, E, E2);
    k_scatter<<<bEC, 256, 0, stream>>>(XL, EB, DEN, ei, E, E2, AGG);
    k_post   <<<bN, 256, 0, stream>>>(AGG, bias2, g2, be2, m2, v2, H2, H, N);

    // ---- output head ----
    float* T = XL;  // reuse: N x 64
    gemm_wmma<2, HID, 64><<<gRow, 128, 0, stream>>>(H, N, Wo1, bo1, 0, 0, 0, 0, T);
    k_gemv<<<bGV, 256, 0, stream>>>(T, Wo2, bo2, (float*)d_out, N);
}
